// Face3DMMOneHotFormer_42623255445990
// MI455X (gfx1250) — matrix operand order repaired and checked by reading, not compiled, not run on hardware
//
#include <hip/hip_runtime.h>
#include <hip/hip_bf16.h>
#include <math.h>

// Face3DMMOneHotFormer for gfx1250 (MI455X), wave32 + WMMA bf16.
//
// Key simplifications (mathematically exact in f32):
//  * causal self-attention + scan  ==> incremental decode with KV cache
//  * diagonal mem_bias (-1e9 off-diag) ==> cross-attn output == V_ca row,
//    so the whole CA block is a precomputable per-row constant.
//  * final decode row i == row i computed at step i; d_out written per step.
//
// One persistent kernel (16 blocks x 256 threads = 128 waves) runs the whole
// 64-step sequence with atomic grid barriers; all GEMVs use
// v_wmma_f32_16x16x32_bf16 (weights converted to bf16 once per launch).
// Weights (~21 MB bf16) are fully L2-resident; an inline-asm WGP-scope
// global_prefetch_b8 (scope 0: "pull into all cache levels") moves the weight
// stream into WGP$ ahead of the pipelined b128 loads.

#define T_SEQ   64
#define DMODEL  1024
#define NHEAD   4
#define HDIM    256
#define DFF     2048
#define PERIOD  30

#define GRID_BLOCKS   16
#define BLOCK_THREADS 256

typedef __attribute__((ext_vector_type(16))) __bf16 v16bf;
typedef __attribute__((ext_vector_type(8)))  __bf16 v8bf;
typedef __attribute__((ext_vector_type(8)))  float  v8f;

union FragBF { v16bf v; v8bf h[2]; };

// ---------------- workspace layout (bytes) ----------------
static constexpr size_t OFF_BAR    = 0;                             // 16KB barrier counters
static constexpr size_t OFF_SAINW  = 16384;                         // bf16 weights
static constexpr size_t OFF_SAOUTW = OFF_SAINW  + 3ull*1024*1024*2;
static constexpr size_t OFF_FF1W   = OFF_SAOUTW + 1ull*1024*1024*2;
static constexpr size_t OFF_FF2W   = OFF_FF1W   + 2ull*1024*1024*2;
static constexpr size_t OFF_VRW    = OFF_FF2W   + 2ull*1024*1024*2;
static constexpr size_t OFF_VMW    = OFF_VRW    + 1ull*1024*1024*2;
static constexpr size_t OFF_PE     = OFF_VMW    + 1ull*1024*1024*2; // f32 scratch
static constexpr size_t OFF_EMB    = OFF_PE  + 64*1024*4;
static constexpr size_t OFF_KC     = OFF_EMB + 64*1024*4;
static constexpr size_t OFF_VC     = OFF_KC  + 64*1024*4;
static constexpr size_t OFF_MEM    = OFF_VC  + 64*1024*4;
static constexpr size_t OFF_CAV    = OFF_MEM + 64*1024*4;
static constexpr size_t OFF_CAC    = OFF_CAV + 64*1024*4;
static constexpr size_t OFF_QKV    = OFF_CAC + 64*1024*4;   // 3072 f32
static constexpr size_t OFF_XROW   = OFF_QKV + 3072*4;
static constexpr size_t OFF_XBF    = OFF_XROW + 1024*4;
static constexpr size_t OFF_OBF    = OFF_XBF  + 1024*2;
static constexpr size_t OFF_S1     = OFF_OBF  + 1024*2;
static constexpr size_t OFF_X2     = OFF_S1   + 1024*4;
static constexpr size_t OFF_X3     = OFF_X2   + 1024*4;
static constexpr size_t OFF_X3BF   = OFF_X3   + 1024*4;
static constexpr size_t OFF_HBF    = OFF_X3BF + 1024*2;     // 2048 bf16
static constexpr size_t OFF_FROW   = OFF_HBF  + 2048*2;
static constexpr size_t OFF_OUTBF  = OFF_FROW + 1024*4;
static constexpr size_t OFF_RBF    = OFF_OUTBF+ 1024*2;
static constexpr size_t OFF_STYLE  = OFF_RBF  + 1024*2;
static constexpr size_t WS_NEEDED  = OFF_STYLE + 1024*4;    // ~22.9 MB

// ---------------- grid-wide barrier (use-once counters) ----------------
__device__ __forceinline__ void gsync(unsigned* bars, unsigned idx, unsigned nblk) {
  __syncthreads();
  if (threadIdx.x == 0) {
    __threadfence();
    __hip_atomic_fetch_add(&bars[idx], 1u, __ATOMIC_RELEASE, __HIP_MEMORY_SCOPE_AGENT);
    while (__hip_atomic_load(&bars[idx], __ATOMIC_ACQUIRE, __HIP_MEMORY_SCOPE_AGENT) < nblk)
      __builtin_amdgcn_s_sleep(1);
  }
  __syncthreads();
}

// ---------------- WMMA bf16 GEMV: out[n] = sum_k x[k]*W[n,k] (+bias...) ----
// W row-major (N,K) in bf16. A-fragment broadcasts x to all 16 M rows so
// every lane's D element 0 holds out[n0 + lane%16].
// MODE bits: 1=store f32, 2=store bf16, 4=relu, 8=add addv[n]
template<int MODE>
__device__ __forceinline__ void gemv_wmma(
    const __bf16* __restrict__ Wb, const float* __restrict__ bias,
    const __bf16* __restrict__ xbf, int N, int K,
    float* __restrict__ outf, __bf16* __restrict__ outb,
    const float* __restrict__ addv, int gwave, int nwaves, int lane)
{
  const int kb = (lane & 16) ? 16 : 0;   // B: lane-group selects K half
  const int ao = (lane & 16) ? 16 : 0;   // A: lane-group K offset
  for (int t = gwave; t * 16 < N; t += nwaves) {
    const int n0 = t << 4;
    const __bf16* wrow = Wb + (size_t)(n0 + (lane & 15)) * K + kb;
    v8f c = {0.f, 0.f, 0.f, 0.f, 0.f, 0.f, 0.f, 0.f};
    for (int k0 = 0; k0 < K; k0 += 32) {
      FragBF a, b;
      a.h[0] = *(const v8bf*)(xbf + k0 + ao);        // K = k0+ao .. +7
      a.h[1] = *(const v8bf*)(xbf + k0 + ao + 8);    // K = k0+ao+8 .. +15
      b.h[0] = *(const v8bf*)(wrow + k0);            // K = k0+kb .. +7
      b.h[1] = *(const v8bf*)(wrow + k0 + 8);
      // WGP-scope prefetch (ISA scope 0: pull into all cache levels, stop on
      // WGP hit) -- __builtin_prefetch can only reach SE/SYS scope, which do
      // NOT prefetch into WGP$. 512B ahead; address stays inside ws (weights
      // are followed by f32 scratch), safe for non-speculative WGP prefetch.
      asm volatile("global_prefetch_b8 %0, off offset:512"
                   :: "v"((const void*)(wrow + k0)));
      c = __builtin_amdgcn_wmma_f32_16x16x32_bf16(false, a.v, false, b.v,
                                                  (short)0, c, false, false);
    }
    if (lane < 16) {
      const int n = n0 + lane;
      float r = c[0] + bias[n];
      if (MODE & 4) r = fmaxf(r, 0.f);
      if (MODE & 8) r += addv[n];
      if (MODE & 1) outf[n] = r;
      if (MODE & 2) outb[n] = (__bf16)r;
    }
  }
}

// ---------------- plain f32 GEMM for one-time precompute -----------------
__device__ void gemm_f32(const float* __restrict__ X, const float* __restrict__ W,
                         const float* __restrict__ B, float* __restrict__ O,
                         int M, int N, int K, int gtid, int nth)
{
  for (int idx = gtid; idx < M * N; idx += nth) {
    const int m = idx / N, n = idx - m * N;
    const float* x = X + (size_t)m * K;
    const float* w = W + (size_t)n * K;
    float s = B ? B[n] : 0.f;
    for (int k = 0; k < K; k++) s = fmaf(x[k], w[k], s);
    O[idx] = s;
  }
}

// ---------------- LayerNorm over D=1024, one block (256 thr) -------------
__device__ __forceinline__ void ln_row(const float* __restrict__ xa,
                                       const float* __restrict__ xb,
                                       const float* __restrict__ g,
                                       const float* __restrict__ bb,
                                       float* outf, __bf16* outb)
{
  __shared__ float red[BLOCK_THREADS];
  const int tid = threadIdx.x;
  float loc[4]; float s = 0.f;
#pragma unroll
  for (int r = 0; r < 4; r++) { int d = tid + r * 256; float v = xa[d] + xb[d]; loc[r] = v; s += v; }
  red[tid] = s; __syncthreads();
  for (int o = 128; o > 0; o >>= 1) { if (tid < o) red[tid] += red[tid + o]; __syncthreads(); }
  const float mean = red[0] * (1.f / 1024.f); __syncthreads();
  float vs = 0.f;
#pragma unroll
  for (int r = 0; r < 4; r++) { float t = loc[r] - mean; vs += t * t; }
  red[tid] = vs; __syncthreads();
  for (int o = 128; o > 0; o >>= 1) { if (tid < o) red[tid] += red[tid + o]; __syncthreads(); }
  const float inv = rsqrtf(red[0] * (1.f / 1024.f) + 1e-5f); __syncthreads();
#pragma unroll
  for (int r = 0; r < 4; r++) {
    const int d = tid + r * 256;
    const float y = (loc[r] - mean) * inv * g[d] + bb[d];
    if (outf) outf[d] = y;
    if (outb) outb[d] = (__bf16)y;
  }
}

__device__ __forceinline__ void conv_bf16(const float* __restrict__ src,
                                          __bf16* __restrict__ dst, int n,
                                          int gtid, int nth)
{
  for (int i = gtid; i < n; i += nth) dst[i] = (__bf16)src[i];
}

// =========================== main persistent kernel ======================
__global__ __launch_bounds__(BLOCK_THREADS)
void face3dmm_former_kernel(
    const float* __restrict__ hidden, const float* __restrict__ hid_w,
    const float* __restrict__ hid_b,  const float* __restrict__ obj_w,
    const float* __restrict__ sa_in_w, const float* __restrict__ sa_in_b,
    const float* __restrict__ sa_out_w, const float* __restrict__ sa_out_b,
    const float* __restrict__ ca_in_w, const float* __restrict__ ca_in_b,
    const float* __restrict__ ca_out_w, const float* __restrict__ ca_out_b,
    const float* __restrict__ ln1_g, const float* __restrict__ ln1_b,
    const float* __restrict__ ln2_g, const float* __restrict__ ln2_b,
    const float* __restrict__ ln3_g, const float* __restrict__ ln3_b,
    const float* __restrict__ ff1_w, const float* __restrict__ ff1_b,
    const float* __restrict__ ff2_w, const float* __restrict__ ff2_b,
    const float* __restrict__ vr_w,  const float* __restrict__ vr_b,
    const float* __restrict__ vm_w,  const float* __restrict__ vm_b,
    float* __restrict__ out, char* __restrict__ ws)
{
  unsigned* bars   = (unsigned*)(ws + OFF_BAR);
  __bf16* sa_in_wb = (__bf16*)(ws + OFF_SAINW);
  __bf16* sa_out_wb= (__bf16*)(ws + OFF_SAOUTW);
  __bf16* ff1_wb   = (__bf16*)(ws + OFF_FF1W);
  __bf16* ff2_wb   = (__bf16*)(ws + OFF_FF2W);
  __bf16* vr_wb    = (__bf16*)(ws + OFF_VRW);
  __bf16* vm_wb    = (__bf16*)(ws + OFF_VMW);
  float*  pe       = (float*)(ws + OFF_PE);
  float*  emb      = (float*)(ws + OFF_EMB);
  float*  kc       = (float*)(ws + OFF_KC);
  float*  vc       = (float*)(ws + OFF_VC);
  float*  memb     = (float*)(ws + OFF_MEM);
  float*  cav      = (float*)(ws + OFF_CAV);
  float*  cac      = (float*)(ws + OFF_CAC);
  float*  qkv      = (float*)(ws + OFF_QKV);
  float*  xrow     = (float*)(ws + OFF_XROW);
  __bf16* xbf      = (__bf16*)(ws + OFF_XBF);
  __bf16* obf      = (__bf16*)(ws + OFF_OBF);
  float*  s1       = (float*)(ws + OFF_S1);
  float*  x2       = (float*)(ws + OFF_X2);
  float*  x3       = (float*)(ws + OFF_X3);
  __bf16* x3bf     = (__bf16*)(ws + OFF_X3BF);
  __bf16* hbf      = (__bf16*)(ws + OFF_HBF);
  float*  frow     = (float*)(ws + OFF_FROW);
  __bf16* outbf    = (__bf16*)(ws + OFF_OUTBF);
  __bf16* rbf      = (__bf16*)(ws + OFF_RBF);
  float*  style    = (float*)(ws + OFF_STYLE);

  const int tid    = threadIdx.x;
  const int gtid   = blockIdx.x * BLOCK_THREADS + tid;
  const int nth    = GRID_BLOCKS * BLOCK_THREADS;
  const int lane   = tid & 31;
  const int wave   = tid >> 5;
  const int gwave  = blockIdx.x * (BLOCK_THREADS / 32) + wave;
  const int nwaves = GRID_BLOCKS * (BLOCK_THREADS / 32);
  const unsigned nblk = GRID_BLOCKS;
  unsigned bidx = 0;

  // ---- P0: weight conversion, PE, emb/style init ----
  conv_bf16(sa_in_w,  sa_in_wb,  3 * 1024 * 1024, gtid, nth);
  conv_bf16(sa_out_w, sa_out_wb, 1024 * 1024,     gtid, nth);
  conv_bf16(ff1_w,    ff1_wb,    2 * 1024 * 1024, gtid, nth);
  conv_bf16(ff2_w,    ff2_wb,    2 * 1024 * 1024, gtid, nth);
  conv_bf16(vr_w,     vr_wb,     1024 * 1024,     gtid, nth);
  conv_bf16(vm_w,     vm_wb,     1024 * 1024,     gtid, nth);
  for (int idx = gtid; idx < T_SEQ * DMODEL; idx += nth) {
    const int t = idx >> 10, d = idx & 1023;
    const int c2 = d >> 1;
    const float div = expf((float)(2 * c2) * (-9.210340371976184f / 1024.f));
    const float ang = (float)(t % PERIOD) * div;
    pe[idx] = (d & 1) ? cosf(ang) : sinf(ang);
  }
  for (int idx = gtid; idx < T_SEQ * DMODEL; idx += nth) {
    const int t = idx >> 10, d = idx & 1023;
    const float v = (t == 0) ? obj_w[d * 80] : 0.f;   // style = obj_w[:,0]
    emb[idx] = v;
    if (t == 0) style[d] = v;
  }
  gsync(bars, bidx++, nblk);

  // ---- P1..P3: mem, ca_v, ca_const (one-time f32 GEMMs) ----
  gemm_f32(hidden, hid_w, hid_b, memb, 64, 1024, 1024, gtid, nth);
  gsync(bars, bidx++, nblk);
  gemm_f32(memb, ca_in_w + (size_t)2048 * 1024, ca_in_b + 2048, cav, 64, 1024, 1024, gtid, nth);
  gsync(bars, bidx++, nblk);
  gemm_f32(cav, ca_out_w, ca_out_b, cac, 64, 1024, 1024, gtid, nth);
  gsync(bars, bidx++, nblk);

  // ---- 64 incremental decode steps ----
  for (int i = 0; i < T_SEQ; i++) {
    // S0: x_i = emb[i] + pe[i]
    if (blockIdx.x == 0) {
      for (int d = tid; d < DMODEL; d += BLOCK_THREADS) {
        const float v = emb[i * DMODEL + d] + pe[i * DMODEL + d];
        xrow[d] = v; xbf[d] = (__bf16)v;
      }
    }
    gsync(bars, bidx++, nblk);

    // S1: qkv row = x @ sa_in_w^T + b   (N=3072)
    gemv_wmma<1>(sa_in_wb, sa_in_b, xbf, 3 * DMODEL, DMODEL,
                 qkv, nullptr, nullptr, gwave, nwaves, lane);
    gsync(bars, bidx++, nblk);

    // S2: append KV cache + causal ALiBi attention (block 0)
    if (blockIdx.x == 0) {
      for (int d = tid; d < DMODEL; d += BLOCK_THREADS) {
        kc[i * DMODEL + d] = qkv[DMODEL + d];
        vc[i * DMODEL + d] = qkv[2 * DMODEL + d];
      }
      __threadfence_block();
      __syncthreads();
      if (wave < NHEAD) {
        const int h = wave;
        const float slope = exp2f(-2.0f * (float)(h + 1));   // 0.25^(h+1)
        const float* qh = qkv + h * HDIM;
        const int j0 = lane, j1 = lane + 32;
        float sc0 = -3.0e38f, sc1 = -3.0e38f;
        if (j0 <= i) {
          const float* kh = kc + (size_t)j0 * DMODEL + h * HDIM;
          float d = 0.f;
          for (int e = 0; e < HDIM; e++) d = fmaf(qh[e], kh[e], d);
          sc0 = d * 0.0625f - slope * (float)((i - j0) / PERIOD);
        }
        if (j1 <= i) {
          const float* kh = kc + (size_t)j1 * DMODEL + h * HDIM;
          float d = 0.f;
          for (int e = 0; e < HDIM; e++) d = fmaf(qh[e], kh[e], d);
          sc1 = d * 0.0625f - slope * (float)((i - j1) / PERIOD);
        }
        float mx = fmaxf(sc0, sc1);
        for (int o = 16; o; o >>= 1) mx = fmaxf(mx, __shfl_xor(mx, o, 32));
        const float e0 = (j0 <= i) ? expf(sc0 - mx) : 0.f;
        const float e1 = (j1 <= i) ? expf(sc1 - mx) : 0.f;
        float sm = e0 + e1;
        for (int o = 16; o; o >>= 1) sm += __shfl_xor(sm, o, 32);
        const float inv = 1.f / sm;
        const float a0 = e0 * inv, a1 = e1 * inv;
        float acc[8] = {0, 0, 0, 0, 0, 0, 0, 0};
        for (int j = 0; j <= i; j++) {
          const float aj = __shfl((j < 32) ? a0 : a1, j & 31, 32);
          const float* vh = vc + (size_t)j * DMODEL + h * HDIM + lane * 8;
#pragma unroll
          for (int e = 0; e < 8; e++) acc[e] = fmaf(aj, vh[e], acc[e]);
        }
#pragma unroll
        for (int e = 0; e < 8; e++) obf[h * HDIM + lane * 8 + e] = (__bf16)acc[e];
      }
    }
    gsync(bars, bidx++, nblk);

    // S3: sa_out projection
    gemv_wmma<1>(sa_out_wb, sa_out_b, obf, DMODEL, DMODEL,
                 s1, nullptr, nullptr, gwave, nwaves, lane);
    gsync(bars, bidx++, nblk);

    // S4: LN1(x + sa) -> +ca_const -> LN2 -> x3
    if (blockIdx.x == 0) {
      ln_row(xrow, s1, ln1_g, ln1_b, x2, nullptr);
      __syncthreads();
      ln_row(x2, cac + i * DMODEL, ln2_g, ln2_b, x3, x3bf);
    }
    gsync(bars, bidx++, nblk);

    // S5: h = relu(x3 @ ff1^T + b)  (N=2048)
    gemv_wmma<2 | 4>(ff1_wb, ff1_b, x3bf, DFF, DMODEL,
                     nullptr, hbf, nullptr, gwave, nwaves, lane);
    gsync(bars, bidx++, nblk);

    // S6: f = h @ ff2^T + b  (K=2048)
    gemv_wmma<1>(ff2_wb, ff2_b, hbf, DMODEL, DFF,
                 frow, nullptr, nullptr, gwave, nwaves, lane);
    gsync(bars, bidx++, nblk);

    // S7: out = LN3(x3 + f)
    if (blockIdx.x == 0) ln_row(x3, frow, ln3_g, ln3_b, nullptr, outbf);
    gsync(bars, bidx++, nblk);

    // S8: r = out @ vr^T + vr_b  -> final output row i (+ bf16 for vm)
    gemv_wmma<1 | 2>(vr_wb, vr_b, outbf, DMODEL, DMODEL,
                     out + (size_t)i * DMODEL, rbf, nullptr, gwave, nwaves, lane);
    gsync(bars, bidx++, nblk);

    // S9: emb[i+1] = r @ vm^T + vm_b + style
    if (i < T_SEQ - 1) {
      gemv_wmma<1 | 8>(vm_wb, vm_b, rbf, DMODEL, DMODEL,
                       emb + (size_t)(i + 1) * DMODEL, nullptr, style,
                       gwave, nwaves, lane);
      gsync(bars, bidx++, nblk);
    }
  }
}

extern "C" void kernel_launch(void* const* d_in, const int* in_sizes, int n_in,
                              void* d_out, int out_size, void* d_ws, size_t ws_size,
                              hipStream_t stream)
{
  (void)in_sizes; (void)n_in; (void)out_size; (void)ws_size; // needs ws_size >= ~22.9MB
  // zero the use-once barrier counters (graph-capturable)
  hipMemsetAsync(d_ws, 0, 16384, stream);
  const float* A[26];
  for (int i = 0; i < 26; i++) A[i] = (const float*)d_in[i];
  face3dmm_former_kernel<<<GRID_BLOCKS, BLOCK_THREADS, 0, stream>>>(
      A[0], A[1], A[2], A[3], A[4], A[5], A[6], A[7], A[8], A[9],
      A[10], A[11], A[12], A[13], A[14], A[15], A[16], A[17], A[18], A[19],
      A[20], A[21], A[22], A[23], A[24], A[25],
      (float*)d_out, (char*)d_ws);
}
